// MappingNetwork_19516331393286
// MI455X (gfx1250) — matrix operand order repaired and checked by reading, not compiled, not run on hardware
//
#include <hip/hip_runtime.h>
#include <hip/hip_bf16.h>

typedef __bf16 bf16;
typedef __attribute__((ext_vector_type(16))) __bf16 v16bf;
typedef __attribute__((ext_vector_type(8)))  __bf16 v8bf;
typedef __attribute__((ext_vector_type(8)))  float   v8f;

#define BTOT   16384
#define HIDDEN 512
#define NDOM   10
#define NSTYLE 64
#define MAXTILES 138   // 16384/128 + 10

__device__ __forceinline__ bf16 f2bf(float x) {
    // round-to-nearest-even f32 -> bf16
    union { float f; unsigned u; } v; v.f = x;
    unsigned r = v.u + 0x7FFFu + ((v.u >> 16) & 1u);
    unsigned short h = (unsigned short)(r >> 16);
    return __builtin_bit_cast(bf16, h);
}

// ---------------- weight convert + transpose: f32 [D][K][N] -> bf16 [D][N][K]
__global__ void wconv(const float* __restrict__ src, bf16* __restrict__ dst,
                      int D, int K, int N, int total) {
    for (int idx = blockIdx.x * 256 + threadIdx.x; idx < total; idx += gridDim.x * 256) {
        int n = idx % N;
        int k = (idx / N) % K;
        int d = idx / (N * K);
        dst[((size_t)(d * N + n)) * K + k] = f2bf(src[idx]);
    }
}

// ---------------- layer 0: z[B,16] @ W0[16,512] + b0, ReLU -> bf16 [B,512]
__global__ __launch_bounds__(256)
void l0_kernel(const float* __restrict__ z, const float* __restrict__ W0,
               const float* __restrict__ b0, bf16* __restrict__ out) {
    int idx = blockIdx.x * 256 + threadIdx.x;        // exactly BTOT*HIDDEN threads
    int b = idx >> 9, n = idx & 511;
    float acc = b0[n];
#pragma unroll
    for (int k = 0; k < 16; ++k)
        acc = fmaf(z[b * 16 + k], W0[k * 512 + n], acc);
    out[idx] = f2bf(acc > 0.f ? acc : 0.f);
}

// ---------------- domain bucketing (counting sort) ----------------
__global__ void k_init(int* counts, int* cursors, int* numTiles) {
    int t = threadIdx.x;
    if (t < NDOM) { counts[t] = 0; cursors[t] = 0; }
    if (t == 0) *numTiles = 0;
}
__global__ void k_hist(const int* __restrict__ y, int* counts) {
    int i = blockIdx.x * 256 + threadIdx.x;
    if (i < BTOT) atomicAdd(&counts[y[i]], 1);
}
__global__ void k_plan(const int* __restrict__ counts, int* cursors,
                       int4* tdesc, int* numTiles) {
    if (threadIdx.x == 0 && blockIdx.x == 0) {
        int off = 0, nt = 0;
        for (int d = 0; d < NDOM; ++d) {
            cursors[d] = off;
            int c = counts[d];
            for (int t = 0; t < c; t += 128) {
                int rows = c - t; if (rows > 128) rows = 128;
                tdesc[nt++] = make_int4(d, off + t, rows, 0);
            }
            off += c;
        }
        *numTiles = nt;
    }
}
__global__ void k_scatter(const int* __restrict__ y, int* cursors, int* perm) {
    int i = blockIdx.x * 256 + threadIdx.x;
    if (i < BTOT) {
        int pos = atomicAdd(&cursors[y[i]], 1);
        perm[pos] = i;
    }
}

// ---------------- LDS double-buffered WMMA GEMM ----------------
// C = [ReLU](A[128,512] x W[512,BN] + bias)
// A: bf16 row-major [*,512].  Wt: bf16 transposed [ (d*NTOT + n) * 512 + k ].
// MODE 0: trunk          (rows = blockIdx.x*128, identity, store bf16)
// MODE 1: head layer 1   (A rows gathered via perm[p0+r], store bf16 at p0+r)
// MODE 2: head mid layer (A rows = p0+r, store bf16 at p0+r)
// MODE 3: head output    (N=64, no ReLU, f32 scatter to Ofp[perm[p0+r]*64+n])
template <int BN, int NTOT, int MODE>
__global__ __launch_bounds__(256)
void gemm_wmma(const bf16* __restrict__ A, const bf16* __restrict__ Wt,
               const float* __restrict__ bias, bf16* __restrict__ Obf,
               float* __restrict__ Ofp, const int* __restrict__ perm,
               const int4* __restrict__ tdesc, const int* __restrict__ numTiles) {
    int d = 0, p0, rows;
    if (MODE == 0) {
        p0 = blockIdx.x * 128; rows = 128;
    } else {
        if ((int)blockIdx.x >= *numTiles) return;       // block-uniform
        int4 t = tdesc[blockIdx.x];
        d = t.x; p0 = t.y; rows = t.z;
    }
    constexpr int KC  = 64;        // K-chunk per stage (16 WMMAs per barrier)
    constexpr int LDR = 72;        // padded LDS row stride: 36-bank stride, 16B-aligned rows
    __shared__ bf16 As[2][128 * LDR];
    __shared__ bf16 Bs[2][BN * LDR];

    const int tid   = threadIdx.x;
    const int wid   = tid >> 5, lane = tid & 31;
    const int half  = lane >> 4, l15 = lane & 15;
    const int waveM = wid & 3,  waveN = wid >> 2;       // 4x2 wave grid
    constexpr int NS = BN / 32;                          // 16-col subtiles per wave
    const int n0 = blockIdx.y * BN + waveN * (BN / 2);
    const int n0blk = blockIdx.y * BN;

    // ---- cooperative loader: thread owns a 64B half-row of one tile row per chunk
    const int ldRow = tid >> 1;            // 0..127
    const int ldOff = (tid & 1) * 32;      // element offset within 64-wide K chunk
    int rA  = ldRow;
    int rAc = rA < rows ? rA : rows - 1;
    int arowLd;
    if (MODE == 0)      arowLd = p0 + rA;
    else if (MODE == 1) arowLd = perm[p0 + rAc];
    else                arowLd = p0 + rAc;
    const bf16* aLd = A + (size_t)arowLd * HIDDEN + ldOff;
    const bf16* bLd = Wt + (size_t)(d * NTOT + n0blk + (ldRow < BN ? ldRow : 0)) * HIDDEN + ldOff;
    constexpr bool bAlways = (2 * BN >= 256);            // BN=128: all 256 threads load B
    const bool bAct = bAlways || (tid < 2 * BN);
    bf16* aSt = &As[0][ldRow * LDR + ldOff];
    bf16* bSt = &Bs[0][ldRow * LDR + ldOff];

    // ---- per-wave fragment LDS base addresses
    const bf16* aFrag[2];
#pragma unroll
    for (int s = 0; s < 2; ++s)
        aFrag[s] = &As[0][(waveM * 32 + s * 16 + l15) * LDR];
    const bf16* bFrag[NS];
#pragma unroll
    for (int t = 0; t < NS; ++t)
        bFrag[t] = &Bs[0][(waveN * (BN / 2) + t * 16 + l15) * LDR];
    constexpr int ABUF = 128 * LDR;     // buffer stride in elements
    constexpr int BBUF = BN * LDR;

    const v8f zero = {0.f, 0.f, 0.f, 0.f, 0.f, 0.f, 0.f, 0.f};
    v8f acc[2][NS];
#pragma unroll
    for (int s = 0; s < 2; ++s)
#pragma unroll
        for (int t = 0; t < NS; ++t) acc[s][t] = zero;

    // ---- prologue: stage chunk 0 into buffer 0
    v8bf ga[4], gb[4];
#pragma unroll
    for (int i = 0; i < 4; ++i) ga[i] = *(const v8bf*)(aLd + 8 * i);
    if (bAct) {
#pragma unroll
        for (int i = 0; i < 4; ++i) gb[i] = *(const v8bf*)(bLd + 8 * i);
    }
#pragma unroll
    for (int i = 0; i < 4; ++i) *(v8bf*)(aSt + 8 * i) = ga[i];
    if (bAct) {
#pragma unroll
        for (int i = 0; i < 4; ++i) *(v8bf*)(bSt + 8 * i) = gb[i];
    }
    __syncthreads();

    constexpr int NCHUNK = HIDDEN / KC;   // 8
#pragma unroll 1
    for (int c = 0; c < NCHUNK; ++c) {
        const int cur = c & 1, nxt = cur ^ 1;
        // issue global loads for next chunk (latency hidden behind WMMA)
        if (c + 1 < NCHUNK) {
            const int kk = (c + 1) * KC;
#pragma unroll
            for (int i = 0; i < 4; ++i) ga[i] = *(const v8bf*)(aLd + kk + 8 * i);
            if (bAct) {
#pragma unroll
                for (int i = 0; i < 4; ++i) gb[i] = *(const v8bf*)(bLd + kk + 8 * i);
            }
        }
        // compute current chunk from LDS: 2 k-steps of 32, 16 WMMAs
#pragma unroll
        for (int ks = 0; ks < KC; ks += 32) {
            v16bf a[2], b[NS];
#pragma unroll
            for (int s = 0; s < 2; ++s) {
                // A 16x32 layout: lanes<16 hold K 0..7 / 16..23, lanes>=16 hold 8..15 / 24..31
                v8bf lo = *(const v8bf*)(aFrag[s] + cur * ABUF + ks + 8 * half);
                v8bf hi = *(const v8bf*)(aFrag[s] + cur * ABUF + ks + 16 + 8 * half);
                a[s] = __builtin_shufflevector(lo, hi, 0,1,2,3,4,5,6,7,8,9,10,11,12,13,14,15);
            }
#pragma unroll
            for (int t = 0; t < NS; ++t) {
                // B 32x16 layout: lanes<16 hold K 0..7 / 16..23, lanes>=16 hold 8..15 / 24..31
                v8bf lo = *(const v8bf*)(bFrag[t] + cur * BBUF + ks + 8 * half);
                v8bf hi = *(const v8bf*)(bFrag[t] + cur * BBUF + ks + 16 + 8 * half);
                b[t] = __builtin_shufflevector(lo, hi, 0,1,2,3,4,5,6,7,8,9,10,11,12,13,14,15);
            }
#pragma unroll
            for (int s = 0; s < 2; ++s)
#pragma unroll
                for (int t = 0; t < NS; ++t)
                    acc[s][t] = __builtin_amdgcn_wmma_f32_16x16x32_bf16(
                        false, a[s], false, b[t], (short)0, acc[s][t], false, false);
        }
        // stage next chunk into the other buffer
        if (c + 1 < NCHUNK) {
#pragma unroll
            for (int i = 0; i < 4; ++i) *(v8bf*)(aSt + nxt * ABUF + 8 * i) = ga[i];
            if (bAct) {
#pragma unroll
                for (int i = 0; i < 4; ++i) *(v8bf*)(bSt + nxt * BBUF + 8 * i) = gb[i];
            }
        }
        __syncthreads();
    }

    // ---- epilogue: C/D layout -> VGPR i holds row half*8+i, col = lane&15
#pragma unroll
    for (int t = 0; t < NS; ++t) {
        int cn = n0 + t * 16 + l15;
        float bv = bias[d * NTOT + cn];
#pragma unroll
        for (int s = 0; s < 2; ++s) {
#pragma unroll
            for (int i = 0; i < 8; ++i) {
                int rloc = waveM * 32 + s * 16 + half * 8 + i;
                if (rloc < rows) {
                    float v = acc[s][t][i] + bv;
                    if (MODE != 3) v = v > 0.f ? v : 0.f;
                    if (MODE == 3) {
                        int orow = perm[p0 + rloc];
                        Ofp[(size_t)orow * NSTYLE + cn] = v;
                    } else {
                        Obf[(size_t)(p0 + rloc) * NTOT + cn] = f2bf(v);
                    }
                }
            }
        }
    }
}

extern "C" void kernel_launch(void* const* d_in, const int* in_sizes, int n_in,
                              void* d_out, int out_size, void* d_ws, size_t ws_size,
                              hipStream_t stream) {
    const float* z   = (const float*)d_in[0];
    const int*   y   = (const int*)  d_in[1];
    const float* W0  = (const float*)d_in[2];
    const float* b0  = (const float*)d_in[3];
    const float* W1  = (const float*)d_in[4];
    const float* b1  = (const float*)d_in[5];
    const float* W2  = (const float*)d_in[6];
    const float* b2  = (const float*)d_in[7];
    const float* W3  = (const float*)d_in[8];
    const float* b3  = (const float*)d_in[9];
    const float* Wu1 = (const float*)d_in[10];
    const float* bu1 = (const float*)d_in[11];
    const float* Wu2 = (const float*)d_in[12];
    const float* bu2 = (const float*)d_in[13];
    const float* Wu3 = (const float*)d_in[14];
    const float* bu3 = (const float*)d_in[15];
    const float* Wo  = (const float*)d_in[16];
    const float* bo  = (const float*)d_in[17];
    float* out = (float*)d_out;

    // workspace carve-up (256B aligned)
    char* base = (char*)d_ws;
    size_t off = 0;
    auto take = [&](size_t bytes) -> void* {
        void* p = base + off;
        off += (bytes + 255) & ~(size_t)255;
        return p;
    };
    bf16* Wt1  = (bf16*)take((size_t)512 * 512 * 2);
    bf16* Wt2  = (bf16*)take((size_t)512 * 512 * 2);
    bf16* Wt3  = (bf16*)take((size_t)512 * 512 * 2);
    bf16* WtU1 = (bf16*)take((size_t)NDOM * 512 * 512 * 2);
    bf16* WtU2 = (bf16*)take((size_t)NDOM * 512 * 512 * 2);
    bf16* WtU3 = (bf16*)take((size_t)NDOM * 512 * 512 * 2);
    bf16* WtO  = (bf16*)take((size_t)NDOM * 64 * 512 * 2);
    bf16* actA = (bf16*)take((size_t)BTOT * HIDDEN * 2);
    bf16* actB = (bf16*)take((size_t)BTOT * HIDDEN * 2);
    int*  perm    = (int*) take((size_t)BTOT * 4);
    int*  counts  = (int*) take(64);
    int*  cursors = (int*) take(64);
    int*  numTiles= (int*) take(64);
    int4* tdesc   = (int4*)take((size_t)(MAXTILES + 2) * sizeof(int4));

    // 1) weights -> bf16 transposed [d][n][k]
    wconv<<<1024, 256, 0, stream>>>(W1,  Wt1,  1, 512, 512, 512 * 512);
    wconv<<<1024, 256, 0, stream>>>(W2,  Wt2,  1, 512, 512, 512 * 512);
    wconv<<<1024, 256, 0, stream>>>(W3,  Wt3,  1, 512, 512, 512 * 512);
    wconv<<<4096, 256, 0, stream>>>(Wu1, WtU1, NDOM, 512, 512, NDOM * 512 * 512);
    wconv<<<4096, 256, 0, stream>>>(Wu2, WtU2, NDOM, 512, 512, NDOM * 512 * 512);
    wconv<<<4096, 256, 0, stream>>>(Wu3, WtU3, NDOM, 512, 512, NDOM * 512 * 512);
    wconv<<<1024, 256, 0, stream>>>(Wo,  WtO,  NDOM, 512, 64,  NDOM * 512 * 64);

    // 2) trunk
    l0_kernel<<<(BTOT * HIDDEN) / 256, 256, 0, stream>>>(z, W0, b0, actA);
    dim3 gt(BTOT / 128, HIDDEN / 128);
    gemm_wmma<128, 512, 0><<<gt, 256, 0, stream>>>(actA, Wt1, b1, actB, nullptr, nullptr, nullptr, nullptr);
    gemm_wmma<128, 512, 0><<<gt, 256, 0, stream>>>(actB, Wt2, b2, actA, nullptr, nullptr, nullptr, nullptr);
    gemm_wmma<128, 512, 0><<<gt, 256, 0, stream>>>(actA, Wt3, b3, actB, nullptr, nullptr, nullptr, nullptr);

    // 3) bucket rows by domain (counting sort + 128-row tile plan)
    k_init<<<1, 32, 0, stream>>>(counts, cursors, numTiles);
    k_hist<<<BTOT / 256, 256, 0, stream>>>(y, counts);
    k_plan<<<1, 32, 0, stream>>>(counts, cursors, tdesc, numTiles);
    k_scatter<<<BTOT / 256, 256, 0, stream>>>(y, cursors, perm);

    // 4) only the selected head per sample (10x FLOP reduction vs dense)
    dim3 gh(MAXTILES, HIDDEN / 128);
    gemm_wmma<128, 512, 1><<<gh, 256, 0, stream>>>(actB, WtU1, bu1, actA, nullptr, perm, tdesc, numTiles);
    gemm_wmma<128, 512, 2><<<gh, 256, 0, stream>>>(actA, WtU2, bu2, actB, nullptr, perm, tdesc, numTiles);
    gemm_wmma<128, 512, 2><<<gh, 256, 0, stream>>>(actB, WtU3, bu3, actA, nullptr, perm, tdesc, numTiles);
    dim3 go(MAXTILES, 1);
    gemm_wmma<64, 64, 3><<<go, 256, 0, stream>>>(actA, WtO, bo, nullptr, out, perm, tdesc, numTiles);
}
